// AudioEfficientConformerEncoder_2473901163235
// MI455X (gfx1250) — matrix operand order repaired and checked by reading, not compiled, not run on hardware
//
#include <hip/hip_runtime.h>
#include <hip/hip_bf16.h>

typedef long long LL;
typedef __attribute__((ext_vector_type(16))) __bf16 bf16x16;
typedef __attribute__((ext_vector_type(8)))  __bf16 bf16x8;
typedef __attribute__((ext_vector_type(8)))  float  f32x8;

// ---------------------------------------------------------------------------
// WMMA GEMM: C[b][m][n] = act(A[b][m][:] @ W[:][n] + bias[n]) * alpha + add
//   A: f32 with strides sAm/sAk/sAb. WT: bf16, pre-transposed [Np][Kp]
//   (N-major, K-contiguous, zero-padded to Kp%32==0, Np%128==0).
//   64x128 tile per 256-thread block; 8 wave32 waves each compute 32x32
//   (2x2 WMMA 16x16x32 bf16 sub-tiles, f32 accumulation).
// ---------------------------------------------------------------------------
__global__ __launch_bounds__(256) void k_gemm(
    const float* __restrict__ A, LL sAm, LL sAk, LL sAb,
    const __bf16* __restrict__ WT, LL Kp,
    const float* __restrict__ bias,
    const float* __restrict__ addsrc, LL sRm, LL sRb,
    float* __restrict__ C, LL sCm, LL sCb,
    int M, int N, int K, int act, float alpha)
{
    __shared__ __align__(16) __bf16 As[64][32];    // [m][k]
    __shared__ __align__(16) __bf16 Bs[128][32];   // [n][k]

    const int tid  = threadIdx.x;
    const int lane = tid & 31;
    const int wave = tid >> 5;
    const int r    = lane & 15;
    const int hi   = lane >> 4;
    const int m0   = blockIdx.y * 64;
    const int n0   = blockIdx.x * 128;
    const LL  ab   = (LL)blockIdx.z * sAb;
    const int wm   = wave & 1;    // 2 M sub-blocks of 32
    const int wn   = wave >> 1;   // 4 N sub-blocks of 32

    f32x8 acc00 = {0.f,0.f,0.f,0.f,0.f,0.f,0.f,0.f};
    f32x8 acc01 = acc00, acc10 = acc00, acc11 = acc00;

    const int arow = tid >> 2;          // 0..63
    const int akb  = (tid & 3) << 3;    // 0,8,16,24
    const bool aFastBlk = (sAk == 1) && (m0 + 64 <= M);   // block-uniform

    for (int k0 = 0; k0 < K; k0 += 32) {
        const bool fullK = (k0 + 32 <= K);
        // ---- stage A tile (f32 -> bf16) ----
        if (aFastBlk && fullK) {
            int gm = m0 + arow;
            const float4* ap = (const float4*)(A + ab + (LL)gm * sAm + (k0 + akb));
            float4 f0 = ap[0], f1 = ap[1];
            bf16x8 t;
            t[0] = (__bf16)f0.x; t[1] = (__bf16)f0.y; t[2] = (__bf16)f0.z; t[3] = (__bf16)f0.w;
            t[4] = (__bf16)f1.x; t[5] = (__bf16)f1.y; t[6] = (__bf16)f1.z; t[7] = (__bf16)f1.w;
            *(bf16x8*)(&As[arow][akb]) = t;
            if (k0 + 32 < K)
                __builtin_prefetch(A + ab + (LL)gm * sAm + (k0 + 32 + akb), 0, 1);
        } else {
            int gm  = m0 + arow;
            int gmc = gm < M ? gm : M - 1;
            bf16x8 t;
#pragma unroll
            for (int i = 0; i < 8; ++i) {
                int gk  = k0 + akb + i;
                int gkc = gk < K ? gk : K - 1;
                float v = A[ab + (LL)gmc * sAm + (LL)gkc * sAk];   // clamped, branch-free
                t[i] = (__bf16)((gm < M && gk < K) ? v : 0.f);
            }
            *(bf16x8*)(&As[arow][akb]) = t;
        }
        // ---- stage B tile: pure bf16 copy (padding => no bounds checks) ----
#pragma unroll
        for (int s = 0; s < 2; ++s) {
            int idx = tid + s * 256;        // 0..511
            int n   = idx >> 2;             // 0..127
            int q   = idx & 3;              // 16B chunk within 32-k row
            const uint4* src = (const uint4*)(WT + (LL)(n0 + n) * Kp + k0 + q * 8);
            *(uint4*)(&Bs[n][q * 8]) = *src;
        }
        if (k0 + 32 < K)
            __builtin_prefetch(WT + (LL)(n0 + (tid >> 1)) * Kp + (k0 + 32), 0, 1);
        __syncthreads();

        // ---- A fragments: lane(r,hi) holds M row, K = {hi*8..+7, 16+hi*8..+7} ----
        bf16x16 af0, af1;
        {
            bf16x8 l0 = *(const bf16x8*)(&As[wm * 32 + r][hi * 8]);
            bf16x8 h0 = *(const bf16x8*)(&As[wm * 32 + r][16 + hi * 8]);
            bf16x8 l1 = *(const bf16x8*)(&As[wm * 32 + 16 + r][hi * 8]);
            bf16x8 h1 = *(const bf16x8*)(&As[wm * 32 + 16 + r][16 + hi * 8]);
#pragma unroll
            for (int i = 0; i < 8; ++i) {
                af0[i] = l0[i]; af0[8 + i] = h0[i];
                af1[i] = l1[i]; af1[8 + i] = h1[i];
            }
        }
        // ---- B fragments: lane(r,hi) holds N col, K = {hi*8..+7, 16+hi*8..+7} ----
        bf16x16 bf0, bf1;
        {
            bf16x8 x0 = *(const bf16x8*)(&Bs[wn * 32 + r][hi * 8]);
            bf16x8 x1 = *(const bf16x8*)(&Bs[wn * 32 + r][16 + hi * 8]);
            bf16x8 y0 = *(const bf16x8*)(&Bs[wn * 32 + 16 + r][hi * 8]);
            bf16x8 y1 = *(const bf16x8*)(&Bs[wn * 32 + 16 + r][16 + hi * 8]);
#pragma unroll
            for (int i = 0; i < 8; ++i) {
                bf0[i] = x0[i]; bf0[8 + i] = x1[i];
                bf1[i] = y0[i]; bf1[8 + i] = y1[i];
            }
        }
        acc00 = __builtin_amdgcn_wmma_f32_16x16x32_bf16(false, af0, false, bf0, (short)0, acc00, false, false);
        acc01 = __builtin_amdgcn_wmma_f32_16x16x32_bf16(false, af0, false, bf1, (short)0, acc01, false, false);
        acc10 = __builtin_amdgcn_wmma_f32_16x16x32_bf16(false, af1, false, bf0, (short)0, acc10, false, false);
        acc11 = __builtin_amdgcn_wmma_f32_16x16x32_bf16(false, af1, false, bf1, (short)0, acc11, false, false);
        __syncthreads();
    }

    // ---- epilogue: D layout M = v + 8*hi, N = r ----
    const LL cb = (LL)blockIdx.z * sCb;
    const LL rb = (LL)blockIdx.z * sRb;
#pragma unroll
    for (int mi = 0; mi < 2; ++mi) {
#pragma unroll
        for (int v = 0; v < 8; ++v) {
            int gm = m0 + wm * 32 + mi * 16 + v + hi * 8;
            if (gm >= M) continue;
#pragma unroll
            for (int ni = 0; ni < 2; ++ni) {
                int gn = n0 + wn * 32 + ni * 16 + r;
                if (gn >= N) continue;
                float rv;
                if (mi == 0) rv = (ni == 0) ? acc00[v] : acc01[v];
                else         rv = (ni == 0) ? acc10[v] : acc11[v];
                if (bias) rv += bias[gn];
                if (act == 1) rv = rv / (1.f + __expf(-rv));   // swish
                rv *= alpha;
                if (addsrc) rv += addsrc[rb + (LL)gm * sRm + gn];
                C[cb + (LL)gm * sCm + gn] = rv;
            }
        }
    }
}

// cast + transpose weights: WT[n*Kp + k] = W[k*N + n] (zero-padded)
__global__ void k_cast_wT(const float* __restrict__ W, __bf16* __restrict__ WT,
                          int K, int N, int Kp, LL total)
{
    for (LL i = blockIdx.x * (LL)blockDim.x + threadIdx.x; i < total; i += (LL)gridDim.x * blockDim.x) {
        int k = (int)(i % Kp);
        int n = (int)(i / Kp);
        float v = (n < N && k < K) ? W[(LL)k * N + n] : 0.f;
        WT[i] = (__bf16)v;
    }
}

// ---------------------------------------------------------------------------
// Elementwise / reduction kernels
// ---------------------------------------------------------------------------
__global__ __launch_bounds__(256) void k_layernorm(
    const float* __restrict__ in, const float* __restrict__ g,
    const float* __restrict__ b, float* __restrict__ out, int d)
{
    __shared__ float red[256];
    LL row = blockIdx.x;
    const float* x = in + row * (LL)d;
    float* o = out + row * (LL)d;
    int tid = threadIdx.x;
    float s = 0.f;
    for (int i = tid; i < d; i += 256) s += x[i];
    red[tid] = s; __syncthreads();
    for (int st = 128; st > 0; st >>= 1) { if (tid < st) red[tid] += red[tid + st]; __syncthreads(); }
    float mean = red[0] / (float)d; __syncthreads();
    float v = 0.f;
    for (int i = tid; i < d; i += 256) { float t = x[i] - mean; v += t * t; }
    red[tid] = v; __syncthreads();
    for (int st = 128; st > 0; st >>= 1) { if (tid < st) red[tid] += red[tid + st]; __syncthreads(); }
    float inv = rsqrtf(red[0] / (float)d + 1e-5f); __syncthreads();
    for (int i = tid; i < d; i += 256) o[i] = (x[i] - mean) * inv * g[i] + b[i];
}

__global__ __launch_bounds__(256) void k_softmax(
    const float* __restrict__ in, float* __restrict__ out, int d)
{
    __shared__ float red[256];
    const float* x = in + (LL)blockIdx.x * d;
    float* o = out + (LL)blockIdx.x * d;
    int tid = threadIdx.x;
    float mx = -1e30f;
    for (int i = tid; i < d; i += 256) mx = fmaxf(mx, x[i]);
    red[tid] = mx; __syncthreads();
    for (int st = 128; st > 0; st >>= 1) { if (tid < st) red[tid] = fmaxf(red[tid], red[tid + st]); __syncthreads(); }
    mx = red[0]; __syncthreads();
    float s = 0.f;
    for (int i = tid; i < d; i += 256) s += __expf(x[i] - mx);
    red[tid] = s; __syncthreads();
    for (int st = 128; st > 0; st >>= 1) { if (tid < st) red[tid] += red[tid + st]; __syncthreads(); }
    float inv = 1.f / red[0]; __syncthreads();
    for (int i = tid; i < d; i += 256) o[i] = __expf(x[i] - mx) * inv;
}

__global__ void k_glu(const float* __restrict__ in, float* __restrict__ out, LL rows, int dout)
{
    LL total = rows * (LL)dout;
    for (LL i = blockIdx.x * (LL)blockDim.x + threadIdx.x; i < total; i += (LL)gridDim.x * blockDim.x) {
        LL rr = i / dout; int c = (int)(i % dout);
        float a = in[rr * 2 * dout + c];
        float g = in[rr * 2 * dout + dout + c];
        out[i] = a / (1.f + __expf(-g));
    }
}

__global__ void k_dwconv(const float* __restrict__ in, const float* __restrict__ w,
                         const float* __restrict__ dwb, const float* __restrict__ g,
                         const float* __restrict__ bb, float* __restrict__ out,
                         int B, int L, int Lo, int C, int stride)
{
    LL total = (LL)B * Lo * C;
    for (LL i = blockIdx.x * (LL)blockDim.x + threadIdx.x; i < total; i += (LL)gridDim.x * blockDim.x) {
        int c = (int)(i % C);
        int t = (int)((i / C) % Lo);
        int b = (int)(i / ((LL)Lo * C));
        float s = 0.f;
#pragma unroll
        for (int k = 0; k < 15; ++k) {
            int tt = t * stride + k - 7;
            if (tt >= 0 && tt < L) s += in[((LL)b * L + tt) * C + c] * w[c * 15 + k];
        }
        s += dwb[c];
        s = s * g[c] + bb[c];
        out[i] = s / (1.f + __expf(-s));
    }
}

__global__ void k_subsample(const float* __restrict__ x, const float* __restrict__ w,
                            const float* __restrict__ bias, const float* __restrict__ g,
                            const float* __restrict__ bb, const float* __restrict__ fct,
                            float* __restrict__ out, LL total)
{
    for (LL i = blockIdx.x * (LL)blockDim.x + threadIdx.x; i < total; i += (LL)gridDim.x * blockDim.x) {
        int fo = (int)(i % 40);
        LL rr = i / 40;
        int to = (int)(rr % 1024); rr /= 1024;
        int c  = (int)(rr % 180);
        int b  = (int)(rr / 180);
        float s = 0.f;
#pragma unroll
        for (int ky = 0; ky < 3; ++ky)
#pragma unroll
            for (int kx = 0; kx < 3; ++kx) {
                int ti = 2 * to + ky - 1, fi = 2 * fo + kx - 1;
                if (ti >= 0 && ti < 2048 && fi >= 0 && fi < 80)
                    s += x[((LL)b * 2048 + ti) * 80 + fi] * w[(c * 3 + ky) * 3 + kx];
            }
        s += bias[c];
        s = s * g[c] + bb[c];
        s = s / (1.f + __expf(-s));
        s += fct[b * 180 + c];
        out[i] = s;   // layout (b, c, to, fo)
    }
}

__global__ void k_bcast_add(float* __restrict__ x, const float* __restrict__ e, int B, int L, int d)
{
    LL total = (LL)B * L * d;
    for (LL i = blockIdx.x * (LL)blockDim.x + threadIdx.x; i < total; i += (LL)gridDim.x * blockDim.x) {
        int c = (int)(i % d);
        int b = (int)(i / ((LL)L * d));
        x[i] += e[(LL)b * d + c];
    }
}

__global__ void k_patch_pool(const float* __restrict__ h, float* __restrict__ hp,
                             int B, int L, int Lp, int d)
{
    LL total = (LL)B * Lp * d;
    for (LL i = blockIdx.x * (LL)blockDim.x + threadIdx.x; i < total; i += (LL)gridDim.x * blockDim.x) {
        int c  = (int)(i % d);
        int jp = (int)((i / d) % Lp);
        int b  = (int)(i / ((LL)Lp * d));
        float s = 0.f;
#pragma unroll
        for (int t = 0; t < 3; ++t) {
            int j = jp * 3 + t;
            if (j < L) s += h[((LL)b * L + j) * d + c];
        }
        hp[i] = s * (1.f / 3.f);
    }
}

__global__ void k_repeat_add(float* __restrict__ x, const float* __restrict__ o,
                             int B, int L, int Lp, int d)
{
    LL total = (LL)B * L * d;
    for (LL i = blockIdx.x * (LL)blockDim.x + threadIdx.x; i < total; i += (LL)gridDim.x * blockDim.x) {
        int c = (int)(i % d);
        int t = (int)((i / d) % L);
        int b = (int)(i / ((LL)L * d));
        x[i] += o[((LL)b * Lp + t / 3) * d + c];
    }
}

__global__ void k_sinusoid(float* __restrict__ s, int L, int d, LL total)
{
    for (LL i = blockIdx.x * (LL)blockDim.x + threadIdx.x; i < total; i += (LL)gridDim.x * blockDim.x) {
        int col = (int)(i % d);
        int row = (int)(i / d);
        float pos = (float)(L - 1 - row);
        int k = col >> 1;
        float inv = __expf(-__logf(10000.f) * (float)(2 * k) / (float)d);
        float ang = pos * inv;
        s[i] = (col & 1) ? __cosf(ang) : __sinf(ang);
    }
}

// relative-position attention (one block per (b,h,q); softmax in LDS)
__global__ __launch_bounds__(256) void k_attention(
    const float* __restrict__ Q, const float* __restrict__ K, const float* __restrict__ V,
    const float* __restrict__ E, const float* __restrict__ u, const float* __restrict__ vb,
    const int* __restrict__ plen, int pm, float* __restrict__ O, int L, int d, int dh)
{
    __shared__ float qu[128], qv[128];
    __shared__ float sc[1024];
    __shared__ float red[256];
    int b = blockIdx.z, h = blockIdx.y, q = blockIdx.x;
    int tid = threadIdx.x;
    const float* qrow = Q + ((LL)b * L + q) * d + h * dh;
    for (int c = tid; c < dh; c += 256) { qu[c] = qrow[c] + u[h * dh + c]; qv[c] = qrow[c] + vb[h * dh + c]; }
    __syncthreads();
    int len = plen[b];
    float rs = rsqrtf((float)dh);
    for (int j = tid; j < L; j += 256) {
        const float* kr = K + ((LL)b * L + j) * d + h * dh;
        const float* er = E + ((LL)(L - 1 - q + j)) * d + h * dh;
        float s = 0.f, sb = 0.f;
        for (int c = 0; c < dh; ++c) { s += qu[c] * kr[c]; sb += qv[c] * er[c]; }
        sc[j] = (j * pm < len) ? (s + sb) * rs : -1e9f;
    }
    __syncthreads();
    float mx = -1e30f;
    for (int j = tid; j < L; j += 256) mx = fmaxf(mx, sc[j]);
    red[tid] = mx; __syncthreads();
    for (int st = 128; st > 0; st >>= 1) { if (tid < st) red[tid] = fmaxf(red[tid], red[tid + st]); __syncthreads(); }
    mx = red[0]; __syncthreads();
    float sum = 0.f;
    for (int j = tid; j < L; j += 256) { float e2 = __expf(sc[j] - mx); sc[j] = e2; sum += e2; }
    red[tid] = sum; __syncthreads();
    for (int st = 128; st > 0; st >>= 1) { if (tid < st) red[tid] += red[tid + st]; __syncthreads(); }
    float inv = 1.f / red[0]; __syncthreads();
    for (int c = tid; c < dh; c += 256) {
        float o = 0.f;
        const float* vp = V + (LL)b * L * d + h * dh + c;
        for (int j = 0; j < L; ++j) o += sc[j] * vp[(LL)j * d];
        O[((LL)b * L + q) * d + h * dh + c] = o * inv;
    }
}

__global__ void k_len_update(const int* __restrict__ in, int* __restrict__ out, int B, int div)
{
    int t = threadIdx.x;
    if (t < B) out[t] = (in[t] - 1) / div + 1;
}

__global__ void k_len_float(const int* __restrict__ in, float* __restrict__ out, int B)
{
    int t = threadIdx.x;
    if (t < B) out[t] = (float)in[t];
}

// ---------------------------------------------------------------------------
// Host orchestration
// ---------------------------------------------------------------------------
extern "C" void kernel_launch(void* const* d_in, const int* in_sizes, int n_in,
                              void* d_out, int out_size, void* d_ws, size_t ws_size,
                              hipStream_t stream)
{
    (void)in_sizes; (void)n_in; (void)out_size; (void)ws_size;
    const float* x_in  = (const float*)d_in[0];
    const int*   l_in  = (const int*)d_in[1];
    const float* temb  = (const float*)d_in[2];

    struct Lin { const float* w; const float* b; };
    struct LnP { const float* g; const float* b; };
    int cur = 3;
    auto nxt = [&]() { return (const float*)d_in[cur++]; };
    auto lin = [&]() { Lin l; l.b = nxt(); l.w = nxt(); return l; };   // sorted: b, w
    auto lnp = [&]() { LnP l; l.b = nxt(); l.g = nxt(); return l; };   // sorted: b, g

    struct Blk {
        LnP att_ln, bn, conv_ln, ff1_ln, ff2_ln, ln_out;
        Lin diff, ff1_1, ff1_2, ff2_1, ff2_2, pos, pw1, pw2, res, wk, wo, wq, wv;
        const float *dw, *dwb, *u, *vb;
        bool hasRes;
    } blk[16];

    for (int i = 0; i < 16; ++i) {
        Blk& p = blk[i];
        p.att_ln = lnp(); p.bn = lnp(); p.conv_ln = lnp();
        p.diff = lin(); p.dw = nxt(); p.dwb = nxt();
        p.ff1_1 = lin(); p.ff1_2 = lin(); p.ff1_ln = lnp();
        p.ff2_1 = lin(); p.ff2_2 = lin(); p.ff2_ln = lnp();
        p.ln_out = lnp(); p.pos = lin(); p.pw1 = lin(); p.pw2 = lin();
        p.hasRes = (i == 4 || i == 10);
        if (p.hasRes) p.res = lin(); else { p.res.w = nullptr; p.res.b = nullptr; }
        p.u = nxt(); p.vb = nxt();
        p.wk = lin(); p.wo = lin(); p.wq = lin(); p.wv = lin();
    }
    Lin fc_t = lin();
    Lin head = lin();
    Lin ip1[4], ip2[4];
    for (int j = 0; j < 4; ++j) { ip1[j] = lin(); ip2[j] = lin(); }
    Lin lin_in = lin();
    const float* sub_b = nxt(); LnP sub_bn = lnp(); const float* sub_w = nxt();

    // ---- workspace layout (floats) ----
    float* ws = (float*)d_ws;
    const LL SUBSZ = (LL)16 * 180 * 1024 * 40;    // 117,964,800
    const LL CAP   = (LL)16 * 1024 * 720;         // 11,796,480
    const LL SMSZ  = 1048576;
    float* sub = ws;
    float* bh  = ws;              // scratch (reuses sub region post lin_in)
    float* ba  = ws + CAP;
    float* bb  = ws + 2 * CAP;
    float* bc  = ws + 3 * CAP;
    float* X   = ws + SUBSZ;
    float* Y   = ws + SUBSZ + CAP;
    float* sm  = ws + SUBSZ + 2 * CAP;
    float* fct  = sm;
    float* demb = sm + 4096;
    float* sinb = sm + 16384;
    float* Eb   = sm + 16384 + 400000;
    int*   len0 = (int*)(sm + 16384 + 800000);
    int*   len1 = len0 + 32;
    int*   len2 = len0 + 64;
    __bf16* wpool = (__bf16*)(ws + SUBSZ + 2 * CAP + SMSZ);   // 64M bf16 budget

    auto gsz = [](LL total) { return dim3((unsigned)((total + 255) / 256)); };

    // ---- pre-cast all GEMM weights: bf16, transposed, padded ----
    struct Wt { const __bf16* p; LL Kp; };
    LL woff = 0;
    auto castT = [&](Lin l, int K, int N) {
        int Kp = (K + 31) & ~31;
        int Np = (N + 127) & ~127;
        __bf16* dst = wpool + woff;
        LL total = (LL)Np * Kp;
        woff += total;
        k_cast_wT<<<gsz(total), 256, 0, stream>>>(l.w, dst, K, N, Kp, total);
        Wt w; w.p = dst; w.Kp = (LL)Kp; return w;
    };

    static const int din_[16]    = {180,180,180,180,180, 256,256,256,256,256, 256, 360,360,360,360,360};
    static const int dout_[16]   = {180,180,180,180,256, 256,256,256,256,256, 360, 360,360,360,360,360};
    static const int stride_[16] = {1,1,1,1,2, 1,1,1,1,1, 2, 1,1,1,1,1};
    static const int patch_[16]  = {1,1,1,1,1, 0,0,0,0,0, 0, 0,0,0,0,0};

    struct BW { Wt diff, ff1_1, ff1_2, ff2_1, ff2_2, pos, pw1, pw2, res, wq, wk, wv, wo; } bw[16];
    for (int i = 0; i < 16; ++i) {
        int din = din_[i], dout = dout_[i];
        bw[i].diff  = castT(blk[i].diff, 512, din);
        bw[i].ff1_1 = castT(blk[i].ff1_1, din, 4 * din);
        bw[i].ff1_2 = castT(blk[i].ff1_2, 4 * din, din);
        bw[i].wq    = castT(blk[i].wq, din, din);
        bw[i].wk    = castT(blk[i].wk, din, din);
        bw[i].wv    = castT(blk[i].wv, din, din);
        bw[i].wo    = castT(blk[i].wo, din, din);
        bw[i].pos   = castT(blk[i].pos, din, din);
        bw[i].pw1   = castT(blk[i].pw1, din, 2 * dout);
        bw[i].pw2   = castT(blk[i].pw2, dout, dout);
        if (blk[i].hasRes) bw[i].res = castT(blk[i].res, din, dout);
        bw[i].ff2_1 = castT(blk[i].ff2_1, dout, 4 * dout);
        bw[i].ff2_2 = castT(blk[i].ff2_2, 4 * dout, dout);
    }
    Wt w_fct  = castT(fc_t, 512, 180);
    Wt w_lin  = castT(lin_in, 7200, 180);
    Wt w_head = castT(head, 360, 256);
    Wt w_ip1[4], w_ip2[4];
    static const int interd[4] = {180, 256, 256, 360};
    for (int j = 0; j < 4; ++j) {
        w_ip1[j] = castT(ip1[j], interd[j], 256);
        w_ip2[j] = castT(ip2[j], 256, interd[j]);
    }

    auto gemm = [&](const float* A, LL sAm, LL sAk, LL sAb,
                    Wt wt, const float* bias,
                    const float* add, LL sRm, LL sRb,
                    float* C, LL sCm, LL sCb,
                    int M, int N, int K, int batch, int act, float alpha) {
        dim3 g((N + 127) / 128, (M + 63) / 64, batch);
        k_gemm<<<g, 256, 0, stream>>>(A, sAm, sAk, sAb, wt.p, wt.Kp, bias, add, sRm, sRb,
                                      C, sCm, sCb, M, N, K, act, alpha);
    };

    const int B = 16;
    float* out_head = (float*)d_out;                       // (16,256,256)
    float* out_lens = out_head + (LL)B * 256 * 256;        // (16,)
    float* out_inter[4];
    out_inter[0] = out_lens + B;
    out_inter[1] = out_inter[0] + (LL)B * 1024 * 256;
    out_inter[2] = out_inter[1] + (LL)B * 512 * 256;
    out_inter[3] = out_inter[2] + (LL)B * 512 * 256;

    // ---- frontend ----
    k_len_update<<<1, 32, 0, stream>>>(l_in, len0, B, 2);
    gemm(temb, 512, 1, 0, w_fct, fc_t.b, nullptr, 0, 0, fct, 180, 0, B, 180, 512, 1, 0, 1.f);
    k_subsample<<<gsz(SUBSZ), 256, 0, stream>>>(x_in, sub_w, sub_b, sub_bn.g, sub_bn.b, fct, sub, SUBSZ);
    // lin_in: weird reshape -> A[m=time j][k=feat r] = sub[b*7372800 + r*1024 + j]
    gemm(sub, 1, 1024, (LL)180 * 1024 * 40, w_lin, lin_in.b, nullptr, 0, 0,
         X, 180, (LL)1024 * 180, 1024, 180, 7200, B, 0, 1.f);

    int L = 1024;
    int interJ = 0;
    const int* curlen = len0;

    for (int i = 0; i < 16; ++i) {
        Blk& p = blk[i];
        const int din = din_[i], dout = dout_[i], stride = stride_[i], patch = patch_[i];
        const int dh = din / 4;
        LL rows = (LL)B * L;

        // diffusion-step embed add
        gemm(temb, 512, 1, 0, bw[i].diff, p.diff.b, nullptr, 0, 0, demb, din, 0, B, din, 512, 1, 0, 1.f);
        k_bcast_add<<<gsz(rows * din), 256, 0, stream>>>(X, demb, B, L, din);

        // FFN1: x += 0.5 * W2(swish(W1(ln(x))))
        k_layernorm<<<(unsigned)rows, 256, 0, stream>>>(X, p.ff1_ln.g, p.ff1_ln.b, ba, din);
        gemm(ba, din, 1, 0, bw[i].ff1_1, p.ff1_1.b, nullptr, 0, 0, bh, 4 * din, 0, (int)rows, 4 * din, din, 1, 1, 1.f);
        gemm(bh, 4 * din, 1, 0, bw[i].ff1_2, p.ff1_2.b, X, din, 0, Y, din, 0, (int)rows, din, 4 * din, 1, 0, 0.5f);
        { float* t = X; X = Y; Y = t; }

        // MHSA with relative positions (optionally patch-pooled)
        k_layernorm<<<(unsigned)rows, 256, 0, stream>>>(X, p.att_ln.g, p.att_ln.b, ba, din);
        int Latt = patch ? (L + 2) / 3 : L;
        LL arows = (LL)B * Latt;
        const float* attin; float *q, *kk, *v, *ctx;
        if (patch) {
            k_patch_pool<<<gsz(arows * din), 256, 0, stream>>>(ba, bc, B, L, Latt, din);
            attin = bc; q = bb; kk = bh; v = ba; ctx = bc;
        } else {
            attin = ba; q = bb; kk = bh; v = bc; ctx = ba;
        }
        gemm(attin, din, 1, 0, bw[i].wq, p.wq.b, nullptr, 0, 0, q,  din, 0, (int)arows, din, din, 1, 0, 1.f);
        gemm(attin, din, 1, 0, bw[i].wk, p.wk.b, nullptr, 0, 0, kk, din, 0, (int)arows, din, din, 1, 0, 1.f);
        gemm(attin, din, 1, 0, bw[i].wv, p.wv.b, nullptr, 0, 0, v,  din, 0, (int)arows, din, din, 1, 0, 1.f);
        int R = 2 * Latt - 1;
        k_sinusoid<<<gsz((LL)R * din), 256, 0, stream>>>(sinb, Latt, din, (LL)R * din);
        gemm(sinb, din, 1, 0, bw[i].pos, p.pos.b, nullptr, 0, 0, Eb, din, 0, R, din, din, 1, 0, 1.f);
        {
            dim3 ag(Latt, 4, B);
            k_attention<<<ag, 256, 0, stream>>>(q, kk, v, Eb, p.u, p.vb, curlen,
                                                patch ? 3 : 1, ctx, Latt, din, dh);
        }
        if (patch) {
            gemm(ctx, din, 1, 0, bw[i].wo, p.wo.b, nullptr, 0, 0, bb, din, 0, (int)arows, din, din, 1, 0, 1.f);
            k_repeat_add<<<gsz(rows * din), 256, 0, stream>>>(X, bb, B, L, Latt, din);
        } else {
            gemm(ctx, din, 1, 0, bw[i].wo, p.wo.b, X, din, 0, Y, din, 0, (int)arows, din, din, 1, 0, 1.f);
            { float* t = X; X = Y; Y = t; }
        }

        // Conv module
        k_layernorm<<<(unsigned)rows, 256, 0, stream>>>(X, p.conv_ln.g, p.conv_ln.b, ba, din);
        gemm(ba, din, 1, 0, bw[i].pw1, p.pw1.b, nullptr, 0, 0, bh, 2 * dout, 0, (int)rows, 2 * dout, din, 1, 0, 1.f);
        k_glu<<<gsz(rows * dout), 256, 0, stream>>>(bh, bb, rows, dout);
        int Lo = (stride == 2) ? (L - 1) / 2 + 1 : L;
        LL orows = (LL)B * Lo;
        k_dwconv<<<gsz(orows * dout), 256, 0, stream>>>(bb, p.dw, p.dwb, p.bn.g, p.bn.b, bc, B, L, Lo, dout, stride);
        const float* res; LL sRm;
        if (p.hasRes) {
            gemm(X, (LL)stride * din, 1, 0, bw[i].res, p.res.b, nullptr, 0, 0, ba, dout, 0, (int)orows, dout, din, 1, 0, 1.f);
            res = ba; sRm = dout;
        } else { res = X; sRm = din; }
        gemm(bc, dout, 1, 0, bw[i].pw2, p.pw2.b, res, sRm, 0, Y, dout, 0, (int)orows, dout, dout, 1, 0, 1.f);
        { float* t = X; X = Y; Y = t; }
        L = Lo; rows = orows;

        // FFN2
        k_layernorm<<<(unsigned)rows, 256, 0, stream>>>(X, p.ff2_ln.g, p.ff2_ln.b, ba, dout);
        gemm(ba, dout, 1, 0, bw[i].ff2_1, p.ff2_1.b, nullptr, 0, 0, bh, 4 * dout, 0, (int)rows, 4 * dout, dout, 1, 1, 1.f);
        gemm(bh, 4 * dout, 1, 0, bw[i].ff2_2, p.ff2_2.b, X, dout, 0, Y, dout, 0, (int)rows, dout, 4 * dout, 1, 0, 0.5f);
        { float* t = X; X = Y; Y = t; }

        // final LN (in place)
        k_layernorm<<<(unsigned)rows, 256, 0, stream>>>(X, p.ln_out.g, p.ln_out.b, X, dout);

        // intermediate CTC heads
        if (i == 2 || i == 5 || i == 9 || i == 12) {
            int j = interJ++;
            gemm(X, dout, 1, 0, w_ip1[j], ip1[j].b, nullptr, 0, 0, out_inter[j], 256, 0, (int)rows, 256, dout, 1, 0, 1.f);
            k_softmax<<<(unsigned)rows, 256, 0, stream>>>(out_inter[j], bb, 256);
            gemm(bb, 256, 1, 0, w_ip2[j], ip2[j].b, X, dout, 0, Y, dout, 0, (int)rows, dout, 256, 1, 0, 1.f);
            { float* t = X; X = Y; Y = t; }
        }

        // length downsampling after stride blocks
        if (stride == 2) {
            if (i == 4) { k_len_update<<<1, 32, 0, stream>>>(len0, len1, B, 2); curlen = len1; }
            else        { k_len_update<<<1, 32, 0, stream>>>(len1, len2, B, 2); curlen = len2; }
        }
    }

    // head + lengths
    gemm(X, 360, 1, 0, w_head, head.b, nullptr, 0, 0, out_head, 256, 0, (int)((LL)B * L), 256, 360, 1, 0, 1.f);
    k_len_float<<<1, 32, 0, stream>>>(curlen, out_lens, B);
}
